// AutoAttention_34170759807652
// MI455X (gfx1250) — compile-verified, matrix operand order checked by reading, not run on hardware
//
#include <hip/hip_runtime.h>

typedef __attribute__((ext_vector_type(16))) _Float16     v16h;
typedef __attribute__((ext_vector_type(8)))  float        v8f;
typedef __attribute__((ext_vector_type(4)))  unsigned int u32x4;

union AFrag { v16h h; u32x4 q[2]; };
union Acc8  { v8f  v; float s[8]; };

#define BB 256
#define NN 197
#define CC 768
#define HH 12
#define DD 64
#define NP 256              // padded N (16 WMMA tiles)
#define MM (BB * NN)        // 50432 = 394 * 128

__device__ __forceinline__ v8f wmma16x16x32(const AFrag& a, const AFrag& b, v8f c) {
  return __builtin_amdgcn_wmma_f32_16x16x32_f16(false, a.h, false, b.h, (short)0, c,
                                                false, false);
}

// LDS byte address of a generic pointer to a __shared__ object:
// flat LDS aperture uses the low 32 address bits as the LDS address (CDNA5 ISA 10.2).
__device__ __forceinline__ unsigned lds_addr(const void* p) {
  return (unsigned)(unsigned long long)p;
}

// async copy of 16 bytes global -> LDS (ASYNCcnt-tracked, per-lane)
__device__ __forceinline__ void async_b128(unsigned lds, const void* gaddr) {
  asm volatile("global_load_async_to_lds_b128 %0, %1, off"
               :: "v"(lds), "v"(gaddr) : "memory");
}
__device__ __forceinline__ void wait_async_all() {
  asm volatile("s_wait_asynccnt 0x0" ::: "memory");
}
__device__ __forceinline__ void wait_async_4() {
  asm volatile("s_wait_asynccnt 0x4" ::: "memory");
}

// ---------------------------------------------------------------------------
// f32 -> f16 cast (activations)
// ---------------------------------------------------------------------------
__global__ void cast_half_kernel(const float* __restrict__ src,
                                 _Float16* __restrict__ dst, int n) {
  int i = blockIdx.x * blockDim.x + threadIdx.x;
  if (i < n) dst[i] = (_Float16)src[i];
}

// ---------------------------------------------------------------------------
// Weight cast + transpose: W (K, Ncols) f32 row-major -> Wt (Ncols, K) f16
// ---------------------------------------------------------------------------
__global__ void transpose_cast_kernel(const float* __restrict__ W,
                                      _Float16* __restrict__ Wt,
                                      int K, int Ncols) {
  int i = blockIdx.x * blockDim.x + threadIdx.x;
  if (i >= K * Ncols) return;
  int k = i / Ncols;
  int n = i - k * Ncols;
  Wt[(size_t)n * K + k] = (_Float16)W[i];
}

// ---------------------------------------------------------------------------
// Tiled WMMA GEMM: D = A(M,K)f16 * Bt(Ncols,K)^T f16 + bias
// Double-buffered async-to-LDS staging (4 async b128 per thread per stage).
// MODE 0: scatter into q/k/v buffers (qkv projection)
// MODE 1: f32 output (final projection)
// ---------------------------------------------------------------------------
template<int MODE>
__launch_bounds__(256)
__global__ void gemm_wmma_kernel(const _Float16* __restrict__ Ah,
                                 const _Float16* __restrict__ Bt,
                                 const float* __restrict__ bias,
                                 int Kdim, int Ncols,
                                 _Float16* __restrict__ qh,
                                 _Float16* __restrict__ kh,
                                 _Float16* __restrict__ vt,
                                 float* __restrict__ outp) {
  __shared__ __align__(16) _Float16 As[2][128 * 32];
  __shared__ __align__(16) _Float16 Bs[2][128 * 32];

  const int tid  = threadIdx.x;
  const int lane = tid & 31;
  const int wave = tid >> 5;            // 0..7
  const int wm   = wave >> 2;           // 0..1  (64-row group)
  const int wn   = wave & 3;            // 0..3  (32-col group)
  const int m0   = blockIdx.y * 128;
  const int n0   = blockIdx.x * 128;
  const int lrow = tid >> 1;            // 0..127
  const int lcol = (tid & 1) * 16;      // 0 / 16 (halves)
  const int lr   = lane & 15;
  const int hk   = (lane >> 4) << 3;    // 0 / 8
  const int rowoff = (lane >> 4) << 3;  // C/D layout row offset

  const _Float16* gA0 = Ah + (size_t)(m0 + lrow) * Kdim + lcol;
  const _Float16* gB0 = Bt + (size_t)(n0 + lrow) * Kdim + lcol;
  const unsigned  lA0 = lds_addr(&As[0][lrow * 32 + lcol]);
  const unsigned  lB0 = lds_addr(&Bs[0][lrow * 32 + lcol]);
  const unsigned  bufStride = lds_addr(&As[1][0]) - lds_addr(&As[0][0]);

  auto issue_stage = [&](int k0, int sbuf) {
    const unsigned o = sbuf ? bufStride : 0u;
    async_b128(lA0 + o,      gA0 + k0);
    async_b128(lA0 + o + 16, gA0 + k0 + 8);
    async_b128(lB0 + o,      gB0 + k0);
    async_b128(lB0 + o + 16, gB0 + k0 + 8);
  };

  Acc8 acc[4][2];
#pragma unroll
  for (int i = 0; i < 4; i++)
#pragma unroll
    for (int j = 0; j < 2; j++)
#pragma unroll
      for (int v = 0; v < 8; v++) acc[i][j].s[v] = 0.0f;

  const int NS = Kdim >> 5;             // K stages of 32
  issue_stage(0, 0);

  for (int s = 0; s < NS; s++) {
    const int cur = s & 1;
    if (s + 1 < NS) {
      issue_stage((s + 1) * 32, cur ^ 1);
      wait_async_4();                   // oldest 4 (stage s) have landed
    } else {
      wait_async_all();
    }
    __syncthreads();                    // publish buffer `cur`

    AFrag af[4], bf[2];
#pragma unroll
    for (int i = 0; i < 4; i++) {
      const int r = wm * 64 + i * 16 + lr;
      af[i].q[0] = *(const u32x4*)&As[cur][r * 32 + hk];
      af[i].q[1] = *(const u32x4*)&As[cur][r * 32 + 16 + hk];
    }
#pragma unroll
    for (int j = 0; j < 2; j++) {
      const int r = wn * 32 + j * 16 + lr;
      bf[j].q[0] = *(const u32x4*)&Bs[cur][r * 32 + hk];
      bf[j].q[1] = *(const u32x4*)&Bs[cur][r * 32 + 16 + hk];
    }
#pragma unroll
    for (int i = 0; i < 4; i++)
#pragma unroll
      for (int j = 0; j < 2; j++)
        acc[i][j].v = wmma16x16x32(af[i], bf[j], acc[i][j].v);
    __syncthreads();                    // done reading buffer `cur`
  }

  // ---- epilogue ----
#pragma unroll
  for (int j = 0; j < 2; j++) {
    const int col = n0 + wn * 32 + j * 16 + lr;
    const float bv = bias[col];
#pragma unroll
    for (int i = 0; i < 4; i++) {
#pragma unroll
      for (int v = 0; v < 8; v++) {
        const int m = m0 + wm * 64 + i * 16 + rowoff + v;
        const float val = acc[i][j].s[v] + bv;
        if (MODE == 0) {
          const int s   = col / CC;          // 0=q 1=k 2=v
          const int rem = col - s * CC;
          const int h   = rem >> 6;
          const int d   = rem & 63;
          const int b   = m / NN;
          const int n   = m - b * NN;
          if (s == 0)
            qh[(((size_t)b * HH + h) * NP + n) * DD + d] = (_Float16)val;
          else if (s == 1)
            kh[(((size_t)b * HH + h) * NP + n) * DD + d] = (_Float16)val;
          else
            vt[(((size_t)b * HH + h) * DD + d) * NP + n] = (_Float16)val;
        } else {
          outp[(size_t)m * Ncols + col] = val;
        }
      }
    }
  }
}

// ---------------------------------------------------------------------------
// Attention: one WG (4 waves) per 32 query rows of one (b,h).
// S kept in accumulator registers; softmax via shfl + small LDS; O = P @ V.
// Q/K/V staged with async-to-LDS.
// ---------------------------------------------------------------------------
__launch_bounds__(128)
__global__ void attn_wmma_kernel(const _Float16* __restrict__ qh,
                                 const _Float16* __restrict__ kh,
                                 const _Float16* __restrict__ vt,
                                 _Float16* __restrict__ attnout) {
  __shared__ __align__(16) _Float16 Qs[32 * 64];
  __shared__ __align__(16) _Float16 Ks[64 * 64];
  __shared__ __align__(16) _Float16 Vs[64 * 64];
  __shared__ __align__(16) _Float16 Ps[32 * 256];
  __shared__ float red[4][32];
  __shared__ float rmaxs[32];
  __shared__ float rinvs[32];

  const int tid    = threadIdx.x;
  const int lane   = tid & 31;
  const int wave   = tid >> 5;           // 0..3
  const int lr     = lane & 15;
  const int hk     = (lane >> 4) << 3;
  const int rowoff = (lane >> 4) << 3;
  const int bh     = blockIdx.y;         // b*12 + h
  const int rb     = blockIdx.x;         // 0..7 (32 rows each)

  // ---- async-load Q strip (32x64) ----
  {
    const int row = tid >> 2;
    const int cb  = (tid & 3) * 16;
    const _Float16* src = qh + ((size_t)bh * NP + rb * 32 + row) * DD + cb;
    const unsigned dst = lds_addr(&Qs[row * 64 + cb]);
    async_b128(dst,      src);
    async_b128(dst + 16, src + 8);
  }
  wait_async_all();
  __syncthreads();

  AFrag aq[2][2];
#pragma unroll
  for (int mt = 0; mt < 2; mt++)
#pragma unroll
    for (int kk = 0; kk < 2; kk++) {
      const int r = mt * 16 + lr;
      aq[mt][kk].q[0] = *(const u32x4*)&Qs[r * 64 + kk * 32 + hk];
      aq[mt][kk].q[1] = *(const u32x4*)&Qs[r * 64 + kk * 32 + 16 + hk];
    }

  // ---- S = scale * Q K^T, streamed over 4 chunks of 64 keys ----
  Acc8 acc[2][4];
#pragma unroll
  for (int mt = 0; mt < 2; mt++)
#pragma unroll
    for (int nc = 0; nc < 4; nc++)
#pragma unroll
      for (int v = 0; v < 8; v++) acc[mt][nc].s[v] = 0.0f;

  const int srow = tid >> 1;             // 0..63
  const int scb  = (tid & 1) * 32;       // 0 / 32 halves

  for (int nc = 0; nc < 4; nc++) {
    {
      const _Float16* src = kh + ((size_t)bh * NP + nc * 64 + srow) * DD + scb;
      const unsigned dst = lds_addr(&Ks[srow * 64 + scb]);
#pragma unroll
      for (int e = 0; e < 4; e++) async_b128(dst + e * 16, src + e * 8);
    }
    wait_async_all();
    __syncthreads();
#pragma unroll
    for (int kk = 0; kk < 2; kk++) {
      AFrag bk;
      const int r = wave * 16 + lr;
      bk.q[0] = *(const u32x4*)&Ks[r * 64 + kk * 32 + hk];
      bk.q[1] = *(const u32x4*)&Ks[r * 64 + kk * 32 + 16 + hk];
#pragma unroll
      for (int mt = 0; mt < 2; mt++)
        acc[mt][nc].v = wmma16x16x32(aq[mt][kk], bk, acc[mt][nc].v);
    }
    __syncthreads();
  }

  // ---- scale + mask + row max ----
  const float scale = 0.125f;  // 64^-0.5
#pragma unroll
  for (int mt = 0; mt < 2; mt++) {
#pragma unroll
    for (int v = 0; v < 8; v++) {
      float mx = -3.0e38f;
#pragma unroll
      for (int nc = 0; nc < 4; nc++) {
        const int n = (nc * 4 + wave) * 16 + lr;
        float val = acc[mt][nc].s[v];
        val = (n < NN) ? val * scale : -1.0e30f;
        acc[mt][nc].s[v] = val;
        mx = fmaxf(mx, val);
      }
      mx = fmaxf(mx, __shfl_xor(mx, 1));
      mx = fmaxf(mx, __shfl_xor(mx, 2));
      mx = fmaxf(mx, __shfl_xor(mx, 4));
      mx = fmaxf(mx, __shfl_xor(mx, 8));
      if (lr == 0) red[wave][mt * 16 + rowoff + v] = mx;
    }
  }
  __syncthreads();
  if (tid < 32)
    rmaxs[tid] = fmaxf(fmaxf(red[0][tid], red[1][tid]),
                       fmaxf(red[2][tid], red[3][tid]));
  __syncthreads();

  // ---- exp + row sum ----
#pragma unroll
  for (int mt = 0; mt < 2; mt++) {
#pragma unroll
    for (int v = 0; v < 8; v++) {
      const int rl = mt * 16 + rowoff + v;
      const float rm = rmaxs[rl];
      float sum = 0.0f;
#pragma unroll
      for (int nc = 0; nc < 4; nc++) {
        float e = __expf(acc[mt][nc].s[v] - rm);
        acc[mt][nc].s[v] = e;
        sum += e;
      }
      sum += __shfl_xor(sum, 1);
      sum += __shfl_xor(sum, 2);
      sum += __shfl_xor(sum, 4);
      sum += __shfl_xor(sum, 8);
      if (lr == 0) red[wave][rl] = sum;
    }
  }
  __syncthreads();
  if (tid < 32)
    rinvs[tid] = 1.0f / (red[0][tid] + red[1][tid] + red[2][tid] + red[3][tid]);
  __syncthreads();

  // ---- write P (f16) ----
#pragma unroll
  for (int mt = 0; mt < 2; mt++)
#pragma unroll
    for (int nc = 0; nc < 4; nc++) {
      const int n = (nc * 4 + wave) * 16 + lr;
#pragma unroll
      for (int v = 0; v < 8; v++) {
        const int rl = mt * 16 + rowoff + v;
        Ps[rl * 256 + n] = (_Float16)(acc[mt][nc].s[v] * rinvs[rl]);
      }
    }
  __syncthreads();

  // ---- O = P @ V (V streamed transposed: vt is (b,h,d,n)) ----
  Acc8 oa[2];
#pragma unroll
  for (int mt = 0; mt < 2; mt++)
#pragma unroll
    for (int v = 0; v < 8; v++) oa[mt].s[v] = 0.0f;

  for (int nc2 = 0; nc2 < 4; nc2++) {
    {
      const _Float16* src = vt + ((size_t)bh * DD + srow) * NP + nc2 * 64 + scb;
      const unsigned dst = lds_addr(&Vs[srow * 64 + scb]);
#pragma unroll
      for (int e = 0; e < 4; e++) async_b128(dst + e * 16, src + e * 8);
    }
    wait_async_all();
    __syncthreads();
#pragma unroll
    for (int kk = 0; kk < 2; kk++) {
      AFrag bv;
      const int r = wave * 16 + lr;             // d column of this wave
      bv.q[0] = *(const u32x4*)&Vs[r * 64 + kk * 32 + hk];
      bv.q[1] = *(const u32x4*)&Vs[r * 64 + kk * 32 + 16 + hk];
#pragma unroll
      for (int mt = 0; mt < 2; mt++) {
        AFrag ap;
        const int pr = mt * 16 + lr;
        ap.q[0] = *(const u32x4*)&Ps[pr * 256 + nc2 * 64 + kk * 32 + hk];
        ap.q[1] = *(const u32x4*)&Ps[pr * 256 + nc2 * 64 + kk * 32 + 16 + hk];
        oa[mt].v = wmma16x16x32(ap, bv, oa[mt].v);
      }
    }
    __syncthreads();
  }

  // ---- store O into (B, 197, 768) f16 ----
  const int b = bh / HH;
  const int h = bh - b * HH;
  const int c = h * DD + wave * 16 + lr;
#pragma unroll
  for (int mt = 0; mt < 2; mt++)
#pragma unroll
    for (int v = 0; v < 8; v++) {
      const int nq = rb * 32 + mt * 16 + rowoff + v;
      if (nq < NN)
        attnout[((size_t)b * NN + nq) * CC + c] = (_Float16)oa[mt].s[v];
    }
}

// ---------------------------------------------------------------------------
// Depthwise 3x3 conv on V (grid 14x14, SAME, per channel) + residual add
// into attention output rows 1..196.
// ---------------------------------------------------------------------------
__global__ void dwconv_residual_kernel(const _Float16* __restrict__ vt,
                                       const float* __restrict__ w,
                                       const float* __restrict__ bias,
                                       _Float16* __restrict__ attnout) {
  const int i = blockIdx.x * blockDim.x + threadIdx.x;
  const int total = BB * CC * 196;
  if (i >= total) return;
  const int p = i % 196;
  const int c = (i / 196) % CC;
  const int b = i / (196 * CC);
  const int h = c >> 6;
  const int d = c & 63;
  const int y = p / 14;
  const int x = p - y * 14;
  const _Float16* vbase = vt + (((size_t)b * HH + h) * DD + d) * NP + 1;
  float sum = bias[c];
  const float* wc = w + c * 9;
#pragma unroll
  for (int dy = -1; dy <= 1; dy++) {
#pragma unroll
    for (int dx = -1; dx <= 1; dx++) {
      const int yy = y + dy, xx = x + dx;
      if (yy >= 0 && yy < 14 && xx >= 0 && xx < 14)
        sum += wc[(dy + 1) * 3 + (dx + 1)] * (float)vbase[yy * 14 + xx];
    }
  }
  const size_t oi = ((size_t)b * NN + (p + 1)) * CC + c;
  attnout[oi] = (_Float16)((float)attnout[oi] + sum);
}

// ---------------------------------------------------------------------------
extern "C" void kernel_launch(void* const* d_in, const int* in_sizes, int n_in,
                              void* d_out, int out_size, void* d_ws, size_t ws_size,
                              hipStream_t stream) {
  const float* x     = (const float*)d_in[0];
  const float* Wqkv  = (const float*)d_in[1];
  const float* bqkv  = (const float*)d_in[2];
  const float* Wproj = (const float*)d_in[3];
  const float* bproj = (const float*)d_in[4];
  const float* dwcw  = (const float*)d_in[5];
  const float* dwcb  = (const float*)d_in[6];
  float* out = (float*)d_out;

  char* ws = (char*)d_ws;
  size_t off = 0;
  _Float16* xh     = (_Float16*)(ws + off); off += (size_t)MM * CC * 2;
  _Float16* WqkvT  = (_Float16*)(ws + off); off += (size_t)(3 * CC) * CC * 2;
  _Float16* WprojT = (_Float16*)(ws + off); off += (size_t)CC * CC * 2;
  _Float16* qh     = (_Float16*)(ws + off); off += (size_t)BB * HH * NP * DD * 2;
  _Float16* kh     = (_Float16*)(ws + off); off += (size_t)BB * HH * NP * DD * 2;
  _Float16* vt     = (_Float16*)(ws + off); off += (size_t)BB * HH * DD * NP * 2;
  _Float16* attn   = (_Float16*)(ws + off); off += (size_t)MM * CC * 2;

  // 1) cast activations + weights to f16 (weights transposed K-contiguous)
  {
    const int nx = MM * CC;
    cast_half_kernel<<<(nx + 255) / 256, 256, 0, stream>>>(x, xh, nx);
    const int n1 = CC * 3 * CC;
    transpose_cast_kernel<<<(n1 + 255) / 256, 256, 0, stream>>>(Wqkv, WqkvT, CC, 3 * CC);
    const int n2 = CC * CC;
    transpose_cast_kernel<<<(n2 + 255) / 256, 256, 0, stream>>>(Wproj, WprojT, CC, CC);
  }

  // 2) QKV projection GEMM -> scattered q/k/v (f16)
  {
    dim3 grid((3 * CC) / 128, MM / 128);   // (18, 394)
    gemm_wmma_kernel<0><<<grid, 256, 0, stream>>>(
        xh, WqkvT, bqkv, CC, 3 * CC, qh, kh, vt, nullptr);
  }

  // 3) attention
  {
    dim3 grid(NP / 32, BB * HH);           // (8, 3072)
    attn_wmma_kernel<<<grid, 128, 0, stream>>>(qh, kh, vt, attn);
  }

  // 4) depthwise conv residual
  {
    const int total = BB * CC * 196;
    dwconv_residual_kernel<<<(total + 255) / 256, 256, 0, stream>>>(vt, dwcw, dwcb, attn);
  }

  // 5) output projection GEMM -> f32 d_out
  {
    dim3 grid(CC / 128, MM / 128);         // (6, 394)
    gemm_wmma_kernel<1><<<grid, 256, 0, stream>>>(
        attn, WprojT, bproj, CC, CC, nullptr, nullptr, nullptr, out);
  }
}
